// LSTM_64647847740244
// MI455X (gfx1250) — matrix operand order repaired and checked by reading, not compiled, not run on hardware
//
#include <hip/hip_runtime.h>
#include <hip/hip_bf16.h>

typedef __attribute__((ext_vector_type(16))) __bf16 v16bf;
typedef __attribute__((ext_vector_type(8)))  float  v8f;

#define FEATD 256
#define HIDD  512
#define BATCH 256
#define TSTEPS 512

// ---- workspace layout (bytes) ----
// bias f32[2048] | barrier | xA bf16 (T*16*8*32*16) | Wfrag bf16 (8*128*32*16)
// | Ufrag bf16 (16*128*32*16) | hA0 bf16 (16*16*32*16) | hA1
static constexpr size_t OFF_BIAS = 0;                          // 8192 B
static constexpr size_t OFF_BAR  = 8192;                       // 256 B
static constexpr size_t OFF_XA   = 8448;                       // 67,108,864 B
static constexpr size_t OFF_WF   = OFF_XA + 67108864;          // 1,048,576 B
static constexpr size_t OFF_UF   = OFF_WF + 1048576;           // 2,097,152 B
static constexpr size_t OFF_HA0  = OFF_UF + 2097152;           // 262,144 B
static constexpr size_t OFF_HA1  = OFF_HA0 + 262144;           // 262,144 B

// --- pack x[B,T,F] f32 -> bf16 A-fragment layout: idx=(((t*16+mt)*8+kc)*32+lane)*16+j
// lane<16 holds row mt*16+lane, k = kc*32 + j ; lane>=16 holds same row, k = kc*32+16+j
__global__ void lstm_pack_x(const float* __restrict__ x, __hip_bfloat16* __restrict__ xA) {
    long idx = (long)blockIdx.x * blockDim.x + threadIdx.x;     // 33,554,432 threads
    int j    = (int)(idx & 15);
    int lane = (int)((idx >> 4) & 31);
    int kc   = (int)((idx >> 9) & 7);
    int mt   = (int)((idx >> 12) & 15);
    int t    = (int)(idx >> 16);
    int b = mt * 16 + (lane & 15);
    int k = kc * 32 + ((lane < 16) ? 0 : 16) + j;
    float v = x[((long)b * TSTEPS + t) * FEATD + k];
    xA[idx] = __float2bfloat16(v);
}

// --- pack W (gate order i,f,o,c) -> B-fragment layout with sigma(K) permutation:
// idx = ((kc*128+ct)*32+lane)*16+j ; col = ct*16 + lane%16 ;
// k = kc*32 + ((lane<16)? 0 : 16) + j
__global__ void lstm_pack_w(const float* __restrict__ Wi, const float* __restrict__ Wf,
                            const float* __restrict__ Wo, const float* __restrict__ Wc,
                            __hip_bfloat16* __restrict__ wfr) {
    int idx  = blockIdx.x * blockDim.x + threadIdx.x;           // 524,288 threads
    int j    = idx & 15;
    int lane = (idx >> 4) & 31;
    int ct   = (idx >> 9) & 127;
    int kc   = idx >> 16;                                       // 0..7
    int col  = ct * 16 + (lane & 15);
    int g    = col >> 9;
    int wcol = col & 511;
    int k    = kc * 32 + ((lane < 16) ? 0 : 16) + j;
    const float* Wg = (g == 0) ? Wi : (g == 1) ? Wf : (g == 2) ? Wo : Wc;
    wfr[idx] = __float2bfloat16(Wg[k * HIDD + wcol]);
}

__global__ void lstm_pack_u(const float* __restrict__ Ui, const float* __restrict__ Uf,
                            const float* __restrict__ Uo, const float* __restrict__ Uc,
                            __hip_bfloat16* __restrict__ ufr) {
    int idx  = blockIdx.x * blockDim.x + threadIdx.x;           // 1,048,576 threads
    int j    = idx & 15;
    int lane = (idx >> 4) & 31;
    int ct   = (idx >> 9) & 127;
    int kc   = idx >> 16;                                       // 0..15
    int col  = ct * 16 + (lane & 15);
    int g    = col >> 9;
    int ucol = col & 511;
    int k    = kc * 32 + ((lane < 16) ? 0 : 16) + j;
    const float* Ug = (g == 0) ? Ui : (g == 1) ? Uf : (g == 2) ? Uo : Uc;
    ufr[idx] = __float2bfloat16(Ug[k * HIDD + ucol]);
}

// --- init: zero both h buffers (h0 = 0), build bias vector, zero grid barrier
__global__ void lstm_init(const float* __restrict__ bi, const float* __restrict__ bf,
                          const float* __restrict__ bo, const float* __restrict__ bc,
                          float* __restrict__ bias, __hip_bfloat16* __restrict__ hA0,
                          __hip_bfloat16* __restrict__ hA1, unsigned* __restrict__ bar) {
    int i = blockIdx.x * blockDim.x + threadIdx.x;              // 131,072 threads
    if (i < 131072) {
        __hip_bfloat16 z = __float2bfloat16(0.f);
        hA0[i] = z;
        hA1[i] = z;
    }
    if (i < 2048) {
        const float* bg = (i < 512) ? bi : (i < 1024) ? bf : (i < 1536) ? bo : bc;
        bias[i] = bg[i & 511];
    }
    if (i == 0) *bar = 0u;
}

__device__ __forceinline__ float sigf(float x) { return 1.f / (1.f + __expf(-x)); }
__device__ __forceinline__ float tanhfast(float x) {
    float e2 = __expf(2.f * x);
    return (e2 - 1.f) / (e2 + 1.f);
}

// --- persistent LSTM: 128 blocks x 256 threads; one device-wide barrier per step
__global__ __launch_bounds__(256) void lstm_persist(
    const __hip_bfloat16* __restrict__ xA,
    const __hip_bfloat16* __restrict__ wfr,
    const __hip_bfloat16* __restrict__ ufr,
    const float* __restrict__ bias,
    __hip_bfloat16* __restrict__ hA0,
    __hip_bfloat16* __restrict__ hA1,
    float* __restrict__ out,
    unsigned* __restrict__ bar) {
    __shared__ float gs[4][4][16][16];                 // [gate][mi][row][col] = 16 KB

    const int tid  = threadIdx.x;
    const int lane = tid & 31;
    const int w    = tid >> 5;                         // 0..7
    const int mi   = w & 3;
    const int gp   = w >> 2;                           // 0..1 -> gates {0,1} or {2,3}
    const int bm   = blockIdx.x & 3;                   // 4 blocks over batch (64 rows)
    const int jb   = blockIdx.x >> 2;                  // 0..31 hidden-col tile
    const int mtile = bm * 4 + mi;                     // 0..15 (16 batch rows each)
    const int g0 = gp * 2, g1 = gp * 2 + 1;
    const int ct0 = g0 * 32 + jb;                      // column tile in 2048-dim
    const int ct1 = g1 * 32 + jb;
    const float bias0 = bias[g0 * 512 + jb * 16 + (lane & 15)];
    const float bias1 = bias[g1 * 512 + jb * 16 + (lane & 15)];
    const int rhalf = (lane >> 4) * 8;
    const int ncol  = lane & 15;
    const unsigned nblk = gridDim.x;

    float creg[4] = {0.f, 0.f, 0.f, 0.f};             // cell state, register-resident

    for (int t = 0; t < TSTEPS; ++t) {
        const __hip_bfloat16* hRd = (t & 1) ? hA1 : hA0;
        __hip_bfloat16*       hWr = (t & 1) ? hA0 : hA1;

        v8f acc0 = {};
        v8f acc1 = {};

        // x_t @ W  (K = 256, 8 chunks of 32)
        const __hip_bfloat16* xa_base = xA + ((long)(t * 16 + mtile) * 8) * 512 + lane * 16;
        for (int kc = 0; kc < 8; ++kc) {
            v16bf a  = *(const v16bf*)(xa_base + kc * 512);
            v16bf b0 = *(const v16bf*)(wfr + ((kc * 128 + ct0) * 32 + lane) * 16);
            v16bf b1 = *(const v16bf*)(wfr + ((kc * 128 + ct1) * 32 + lane) * 16);
            acc0 = __builtin_amdgcn_wmma_f32_16x16x32_bf16(false, a, false, b0, (short)0, acc0, false, false);
            acc1 = __builtin_amdgcn_wmma_f32_16x16x32_bf16(false, a, false, b1, (short)0, acc1, false, false);
        }
        // h_{t-1} @ U  (K = 512, 16 chunks of 32)
        const __hip_bfloat16* ha_base = hRd + (mtile * 16) * 512 + lane * 16;
        for (int kc = 0; kc < 16; ++kc) {
            v16bf a  = *(const v16bf*)(ha_base + kc * 512);
            v16bf b0 = *(const v16bf*)(ufr + ((kc * 128 + ct0) * 32 + lane) * 16);
            v16bf b1 = *(const v16bf*)(ufr + ((kc * 128 + ct1) * 32 + lane) * 16);
            acc0 = __builtin_amdgcn_wmma_f32_16x16x32_bf16(false, a, false, b0, (short)0, acc0, false, false);
            acc1 = __builtin_amdgcn_wmma_f32_16x16x32_bf16(false, a, false, b1, (short)0, acc1, false, false);
        }

        // spill gate tiles to LDS (C layout: VGPR r, lanes<16 -> M=r, lanes>=16 -> M=8+r)
        #pragma unroll
        for (int r = 0; r < 8; ++r) {
            gs[g0][mi][rhalf + r][ncol] = acc0[r] + bias0;
            gs[g1][mi][rhalf + r][ncol] = acc1[r] + bias1;
        }
        __syncthreads();

        // fused elementwise: 64 rows x 16 cols = 1024 elems, 4 per thread
        #pragma unroll
        for (int e = 0; e < 4; ++e) {
            int idx = tid + e * 256;
            int rl  = idx >> 4;                        // 0..63 local row
            int nn  = idx & 15;
            int mi2 = rl >> 4;
            int rr  = rl & 15;
            float it = sigf(gs[0][mi2][rr][nn]);
            float ft = sigf(gs[1][mi2][rr][nn]);
            float ot = sigf(gs[2][mi2][rr][nn]);
            float cc = tanhfast(gs[3][mi2][rr][nn]);
            float cn = ft * creg[e] + it * cc;
            creg[e]  = cn;
            float hn = ot * tanhfast(cn);

            int brow = bm * 64 + rl;
            int col  = jb * 16 + nn;
            out[((long)brow * TSTEPS + t) * HIDD + col] = hn;

            // scatter h into bf16 A-fragment layout for next step
            int mt2 = brow >> 4, m16 = brow & 15;
            int kc2 = col >> 5,  kk  = col & 31;
            int half = kk >> 4,  jj  = kk & 15;
            hWr[(((mt2 * 16 + kc2) * 32) + (m16 + 16 * half)) * 16 + jj] = __float2bfloat16(hn);
        }

        // device-wide barrier between timesteps
        __threadfence();
        __syncthreads();
        if (tid == 0) {
            __hip_atomic_fetch_add(bar, 1u, __ATOMIC_ACQ_REL, __HIP_MEMORY_SCOPE_AGENT);
            unsigned target = (unsigned)(t + 1) * nblk;
            while (__hip_atomic_load(bar, __ATOMIC_ACQUIRE, __HIP_MEMORY_SCOPE_AGENT) < target) {
                __builtin_amdgcn_s_sleep(1);
            }
        }
        __syncthreads();
    }
}

extern "C" void kernel_launch(void* const* d_in, const int* in_sizes, int n_in,
                              void* d_out, int out_size, void* d_ws, size_t ws_size,
                              hipStream_t stream) {
    // setup_inputs order: x, W_i,U_i,b_i, W_f,U_f,b_f, W_c,U_c,b_c, W_o,U_o,b_o
    const float* x  = (const float*)d_in[0];
    const float* Wi = (const float*)d_in[1];
    const float* Ui = (const float*)d_in[2];
    const float* bi = (const float*)d_in[3];
    const float* Wf = (const float*)d_in[4];
    const float* Uf = (const float*)d_in[5];
    const float* bf = (const float*)d_in[6];
    const float* Wc = (const float*)d_in[7];
    const float* Uc = (const float*)d_in[8];
    const float* bc = (const float*)d_in[9];
    const float* Wo = (const float*)d_in[10];
    const float* Uo = (const float*)d_in[11];
    const float* bo = (const float*)d_in[12];

    char* ws = (char*)d_ws;
    float*          bias = (float*)(ws + OFF_BIAS);
    unsigned*       bar  = (unsigned*)(ws + OFF_BAR);
    __hip_bfloat16* xA   = (__hip_bfloat16*)(ws + OFF_XA);
    __hip_bfloat16* wfr  = (__hip_bfloat16*)(ws + OFF_WF);
    __hip_bfloat16* ufr  = (__hip_bfloat16*)(ws + OFF_UF);
    __hip_bfloat16* hA0  = (__hip_bfloat16*)(ws + OFF_HA0);
    __hip_bfloat16* hA1  = (__hip_bfloat16*)(ws + OFF_HA1);

    lstm_init  <<<512,    256, 0, stream>>>(bi, bf, bo, bc, bias, hA0, hA1, bar);
    lstm_pack_x<<<131072, 256, 0, stream>>>(x, xA);
    lstm_pack_w<<<2048,   256, 0, stream>>>(Wi, Wf, Wo, Wc, wfr);
    lstm_pack_u<<<4096,   256, 0, stream>>>(Ui, Uf, Uo, Uc, ufr);
    lstm_persist<<<128,   256, 0, stream>>>(xA, wfr, ufr, bias, hA0, hA1,
                                            (float*)d_out, bar);
}
